// UAG_RNN_14035953123343
// MI455X (gfx1250) — matrix operand order repaired and checked by reading, not compiled, not run on hardware
//
#include <hip/hip_runtime.h>
#include <stdint.h>

// Problem dims (B, C, H, W) = (8, 64, 128, 128), fp32 in/out.
#define CH   64
#define HH   128
#define WW   128
#define HWSZ (HH * WW)

typedef __attribute__((ext_vector_type(16))) __bf16 v16bf;
typedef __attribute__((ext_vector_type(8)))  float  v8f;

union Frag {
    uint32_t u[8];
    uint4    q[2];
    v16bf    v;
};

__device__ __forceinline__ uint32_t pk_bf16(float a, float b) {
    union { __bf16 h[2]; uint32_t u; } x;
    x.h[0] = (__bf16)a; x.h[1] = (__bf16)b;
    return x.u;
}

__device__ __forceinline__ float bf16_to_f32(uint16_t h) {
    union { uint32_t u; float f; } x;
    x.u = ((uint32_t)h) << 16;
    return x.f;
}

__device__ __forceinline__ v8f wmma_bf16(const Frag& a, const Frag& b, v8f c) {
    // D(16x16 f32) = A(16x32 bf16) * B(32x16 bf16) + C
    return __builtin_amdgcn_wmma_f32_16x16x32_bf16(
        /*neg_a=*/false, a.v, /*neg_b=*/false, b.v,
        /*c_mod=*/(short)0, c, /*reuse_a=*/false, /*reuse_b=*/false);
}

// B-operand fragment (32x16 bf16) from LDS laid out [pos][ch], pitch 72 u16
// (144 B, 16B-aligned rows). kstep selects K = 32*kstep .. 32*kstep+31.
// Contiguous k order (not the ISA table's): lanes 0-15 hold K 0..15 (N = lane), lanes 16-31 hold
// K 16..31 (N = lane-16) -> one contiguous 32-byte chunk per lane.
__device__ __forceinline__ Frag load_b_frag(const uint16_t* buf, int n,
                                            int kstep, int laneHi) {
    const uint16_t* p = buf + n * 72 + kstep * 32 + laneHi * 16;
    Frag f;
    f.q[0] = *(const uint4*)p;
    f.q[1] = *(const uint4*)(p + 8);
    return f;
}

// A-operand fragment (16x32 bf16) gathered straight from a global f32 weight
// matrix Wk[64][64] (row = c_out, col = c_in), converted to bf16 in regs.
// Contiguous k order, same as load_b_frag: lane<16 holds M=lane, K = 32s+{0..15};
// lane>=16 holds M=lane-16, K = 32s+{16..31}.
__device__ __forceinline__ Frag load_a_frag(const float* Wk, int mBase,
                                            int kstep, int lane) {
    const int laneHi = lane >> 4;
    const float* r = Wk + (mBase + (lane & 15)) * 64 + 32 * kstep + laneHi * 16;
    Frag f;
#pragma unroll
    for (int j = 0; j < 4; ++j) f.u[j]     = pk_bf16(r[2 * j],      r[2 * j + 1]);
#pragma unroll
    for (int j = 0; j < 4; ++j) f.u[4 + j] = pk_bf16(r[8 + 2 * j], r[9 + 2 * j]);
    return f;
}

// Async-copy one 64ch x 64w fp32 row tile (16 KB) into LDS using the CDNA5
// async global->LDS path; 256 lanes * 16 B * 4 issues.
__device__ __forceinline__ void stage_row_async(const float* src, float* stage,
                                                int tid) {
#pragma unroll
    for (int q = 0; q < 4; ++q) {
        const int e  = q * 256 + tid;       // 0..1023
        const int c  = e >> 4;              // channel 0..63
        const int wq = (e & 15) * 4;        // 4-float chunk within the row
        uint64_t ga = (uint64_t)(uintptr_t)(src + (size_t)c * HWSZ + wq);
        uint32_t la = (uint32_t)(uintptr_t)(&stage[c * 64 + wq]);
        asm volatile("global_load_async_to_lds_b128 %0, %1, off"
                     :: "v"(la), "v"(ga) : "memory");
    }
    asm volatile("s_wait_asynccnt 0" ::: "memory");
}

__global__ void zero_kernel(float* __restrict__ out, int n) {
    for (int i = blockIdx.x * blockDim.x + threadIdx.x; i < n;
         i += gridDim.x * blockDim.x)
        out[i] = 0.0f;
}

// ---------------------------------------------------------------------------
// Kernel 1: south + north row scans.  h_i = relu(Wa@x_i + (Wb@h_{i-1})*y_prev)
// Grid: 32 blocks = {dir 0/1} x {batch 0..7} x {w-tile 0/1}; 256 threads.
// Row-scan recurrence is independent per (b,w) column -> no cross-WG sync.
// ---------------------------------------------------------------------------
__global__ __launch_bounds__(256) void row_scan_kernel(
    const float* __restrict__ x, const float* __restrict__ y,
    const float* __restrict__ Wc, const float* __restrict__ bc,
    float* __restrict__ hs, float* __restrict__ hn)
{
    __shared__ __align__(16) float    xStage[CH * 64];   // raw fp32 row tile
    __shared__ __align__(16) uint16_t xB[64 * 72];       // bf16 [w][ch]
    __shared__ __align__(16) uint16_t hB[2][64 * 72];    // bf16 state [w][ch]

    const int bid = blockIdx.x;
    const int dir = bid >> 4;            // 0 = south (top->bottom), 1 = north
    const int b   = (bid >> 1) & 7;
    const int w0  = (bid & 1) * 64;
    const int kA  = dir ? 8 : 0;         // conv_a weight index
    const int kB  = dir ? 9 : 1;         // conv_b weight index
    float* hout   = dir ? hn : hs;

    const int tid    = threadIdx.x;
    const int lane   = tid & 31;
    const int wave   = tid >> 5;
    const int laneHi = lane >> 4;
    const int laneN  = lane & 15;
    const int mBase  = (wave >> 1) * 16;        // c_out tile
    const int nBase0 = (wave & 1) * 32;         // two 16-wide w tiles
    const int mOff   = mBase + laneHi * 8;      // C/D frag: VGPR r -> M=mOff+r

    // Weights live in registers for the whole scan (A operand of every WMMA).
    Frag Aa[2], Ab[2];
#pragma unroll
    for (int s = 0; s < 2; ++s) {
        Aa[s] = load_a_frag(Wc + kA * 4096, mBase, s, lane);
        Ab[s] = load_a_frag(Wc + kB * 4096, mBase, s, lane);
    }
    float biasA[8], biasB[8];
#pragma unroll
    for (int r = 0; r < 8; ++r) {
        biasA[r] = bc[kA * 64 + mOff + r];
        biasB[r] = bc[kB * 64 + mOff + r];
    }

    const float* xb = x    + (size_t)b * CH * HWSZ + w0;
    const float* yb = y    + (size_t)b * CH * HWSZ + w0;
    float*       hb = hout + (size_t)b * CH * HWSZ + w0;

    int cur = 0;

    // Boundary row: h0 = x_row (south) or relu(x_row) (north).
    {
        const int i0 = dir ? (HH - 1) : 0;
        stage_row_async(xb + (size_t)i0 * WW, xStage, tid);
        __syncthreads();
        for (int idx = tid; idx < 64 * 16; idx += 256) {
            const int n  = idx & 63;
            const int k4 = (idx >> 6) * 4;
            float v[4];
#pragma unroll
            for (int jj = 0; jj < 4; ++jj) {
                float t = xStage[(k4 + jj) * 64 + n];
                if (dir) t = fmaxf(t, 0.0f);
                hb[(size_t)(k4 + jj) * HWSZ + (size_t)i0 * WW + n] = t;
                v[jj] = t;
            }
            uint2 p;
            p.x = pk_bf16(v[0], v[1]);
            p.y = pk_bf16(v[2], v[3]);
            *(uint2*)&hB[cur][n * 72 + k4] = p;
        }
        __syncthreads();
    }

    for (int t = 1; t < HH; ++t) {
        const int i  = dir ? (HH - 1 - t) : t;
        const int ip = dir ? (i + 1) : (i - 1);   // previous row (state & y)

        // Stage x row i: async fp32 -> LDS, then convert to bf16 [w][ch].
        stage_row_async(xb + (size_t)i * WW, xStage, tid);
        __syncthreads();
        for (int idx = tid; idx < 64 * 16; idx += 256) {
            const int n  = idx & 63;
            const int k4 = (idx >> 6) * 4;
            uint2 p;
            p.x = pk_bf16(xStage[(k4 + 0) * 64 + n], xStage[(k4 + 1) * 64 + n]);
            p.y = pk_bf16(xStage[(k4 + 2) * 64 + n], xStage[(k4 + 3) * 64 + n]);
            *(uint2*)&xB[n * 72 + k4] = p;
        }
        __syncthreads();

#pragma unroll
        for (int half = 0; half < 2; ++half) {
            const int n = nBase0 + 16 * half + laneN;
            Frag Bx0 = load_b_frag(xB,      n, 0, laneHi);
            Frag Bx1 = load_b_frag(xB,      n, 1, laneHi);
            Frag Bh0 = load_b_frag(hB[cur], n, 0, laneHi);
            Frag Bh1 = load_b_frag(hB[cur], n, 1, laneHi);

            v8f accX = {0.f, 0.f, 0.f, 0.f, 0.f, 0.f, 0.f, 0.f};
            accX = wmma_bf16(Aa[0], Bx0, accX);
            accX = wmma_bf16(Aa[1], Bx1, accX);
            v8f accH = {0.f, 0.f, 0.f, 0.f, 0.f, 0.f, 0.f, 0.f};
            accH = wmma_bf16(Ab[0], Bh0, accH);
            accH = wmma_bf16(Ab[1], Bh1, accH);

            float hv[8];
#pragma unroll
            for (int r = 0; r < 8; ++r) {
                const float yv =
                    yb[(size_t)(mOff + r) * HWSZ + (size_t)ip * WW + n];
                float v = accX[r] + biasA[r] + (accH[r] + biasB[r]) * yv;
                v = fmaxf(v, 0.0f);
                hb[(size_t)(mOff + r) * HWSZ + (size_t)i * WW + n] = v;
                hv[r] = v;
            }
            uint4 p;
            p.x = pk_bf16(hv[0], hv[1]); p.y = pk_bf16(hv[2], hv[3]);
            p.z = pk_bf16(hv[4], hv[5]); p.w = pk_bf16(hv[6], hv[7]);
            *(uint4*)&hB[cur ^ 1][n * 72 + mOff] = p;
        }
        __syncthreads();
        cur ^= 1;
    }
}

// ---------------------------------------------------------------------------
// Kernel 2: one directional column scan (launched 4x, sequentially, so the
// read-modify-write accumulation into d_out is race-free and deterministic).
// c_j = relu(Wa@base_j + (Wb@c_{j-1})*y_prev + gamma*shift((Wt@c_{j-1})*y_prev))
// Grid: 8 blocks (one per batch); each owns the full [64ch x 128h] state.
// ---------------------------------------------------------------------------
__global__ __launch_bounds__(256) void col_scan_kernel(
    const float* __restrict__ base, const float* __restrict__ y,
    const float* __restrict__ Wc, const float* __restrict__ bc,
    const float* __restrict__ gammas, int gIdx,
    int kT, int kA, int kB, int dirSign, int shiftDown,
    float* __restrict__ out)
{
    __shared__ __align__(16) uint16_t prevB[128 * 72];  // state bf16 [h][ch]
    __shared__ __align__(16) uint16_t baseB[128 * 72];  // base col bf16 [h][ch]
    __shared__ __align__(16) uint16_t gbuf[128 * 64];   // gated bf16 [h][ch]

    const int b = blockIdx.x;
    const float gamma = gammas[gIdx];

    const int tid    = threadIdx.x;
    const int lane   = tid & 31;
    const int wave   = tid >> 5;
    const int laneHi = lane >> 4;
    const int laneN  = lane & 15;
    const int mBase  = (wave >> 1) * 16;
    const int nGrp   = (wave & 1) * 4;          // 4 h-tiles per wave
    const int mOff   = mBase + laneHi * 8;

    Frag At[2], Aw[2], Ab2[2];
#pragma unroll
    for (int s = 0; s < 2; ++s) {
        At[s]  = load_a_frag(Wc + kT * 4096, mBase, s, lane);
        Aw[s]  = load_a_frag(Wc + kA * 4096, mBase, s, lane);
        Ab2[s] = load_a_frag(Wc + kB * 4096, mBase, s, lane);
    }
    float bT[8], bA[8], bB[8];
#pragma unroll
    for (int r = 0; r < 8; ++r) {
        bT[r] = bc[kT * 64 + mOff + r];
        bA[r] = bc[kA * 64 + mOff + r];
        bB[r] = bc[kB * 64 + mOff + r];
    }

    const float* bp = base + (size_t)b * CH * HWSZ;
    const float* yp = y    + (size_t)b * CH * HWSZ;
    float*       op = out  + (size_t)b * CH * HWSZ;

    const int j0 = (dirSign > 0) ? 0 : (WW - 1);

    // Init column: state = relu(base[:, :, j0]); it also contributes to out.
    for (int idx = tid; idx < 128 * 16; idx += 256) {
        const int n  = idx & 127;
        const int k4 = (idx >> 7) * 4;
        float v[4];
#pragma unroll
        for (int jj = 0; jj < 4; ++jj) {
            float t = bp[(size_t)(k4 + jj) * HWSZ + (size_t)n * WW + j0];
            t = fmaxf(t, 0.0f);
            op[(size_t)(k4 + jj) * HWSZ + (size_t)n * WW + j0] += t;
            v[jj] = t;
        }
        uint2 p;
        p.x = pk_bf16(v[0], v[1]);
        p.y = pk_bf16(v[2], v[3]);
        *(uint2*)&prevB[n * 72 + k4] = p;
    }
    __syncthreads();

    for (int t = 1; t < WW; ++t) {
        const int j  = j0 + dirSign * t;
        const int jp = j - dirSign;

        // Stage base column j as bf16 [h][ch].
        for (int idx = tid; idx < 128 * 16; idx += 256) {
            const int n  = idx & 127;
            const int k4 = (idx >> 7) * 4;
            uint2 p;
            p.x = pk_bf16(bp[(size_t)(k4 + 0) * HWSZ + (size_t)n * WW + j],
                          bp[(size_t)(k4 + 1) * HWSZ + (size_t)n * WW + j]);
            p.y = pk_bf16(bp[(size_t)(k4 + 2) * HWSZ + (size_t)n * WW + j],
                          bp[(size_t)(k4 + 3) * HWSZ + (size_t)n * WW + j]);
            *(uint2*)&baseB[n * 72 + k4] = p;
        }
        __syncthreads();

        // Phase A: three GEMMs per tile; publish gated term to LDS.
        float part[4][8];
#pragma unroll
        for (int u = 0; u < 4; ++u) {
            const int n = 16 * (nGrp + u) + laneN;
            Frag Bp0 = load_b_frag(prevB, n, 0, laneHi);
            Frag Bp1 = load_b_frag(prevB, n, 1, laneHi);
            Frag Bb0 = load_b_frag(baseB, n, 0, laneHi);
            Frag Bb1 = load_b_frag(baseB, n, 1, laneHi);

            v8f aT = {0.f, 0.f, 0.f, 0.f, 0.f, 0.f, 0.f, 0.f};
            aT = wmma_bf16(At[0], Bp0, aT);
            aT = wmma_bf16(At[1], Bp1, aT);
            v8f aB = {0.f, 0.f, 0.f, 0.f, 0.f, 0.f, 0.f, 0.f};
            aB = wmma_bf16(Ab2[0], Bp0, aB);
            aB = wmma_bf16(Ab2[1], Bp1, aB);
            v8f aA = {0.f, 0.f, 0.f, 0.f, 0.f, 0.f, 0.f, 0.f};
            aA = wmma_bf16(Aw[0], Bb0, aA);
            aA = wmma_bf16(Aw[1], Bb1, aA);

            float g[8];
#pragma unroll
            for (int r = 0; r < 8; ++r) {
                const float yv =
                    yp[(size_t)(mOff + r) * HWSZ + (size_t)n * WW + jp];
                g[r] = (aT[r] + bT[r]) * yv;
                part[u][r] = aA[r] + bA[r] + (aB[r] + bB[r]) * yv;
            }
            uint4 p;
            p.x = pk_bf16(g[0], g[1]); p.y = pk_bf16(g[2], g[3]);
            p.z = pk_bf16(g[4], g[5]); p.w = pk_bf16(g[6], g[7]);
            *(uint4*)&gbuf[n * 64 + mOff] = p;
        }
        __syncthreads();

        // Phase B: read the H-shifted gated term, combine, relu, accumulate.
#pragma unroll
        for (int u = 0; u < 4; ++u) {
            const int n  = 16 * (nGrp + u) + laneN;
            const int ns = shiftDown ? (n - 1) : (n + 1);
            float g[8];
            if (ns >= 0 && ns < HH) {
                uint4 p = *(const uint4*)&gbuf[ns * 64 + mOff];
                g[0] = bf16_to_f32((uint16_t)(p.x & 0xffff));
                g[1] = bf16_to_f32((uint16_t)(p.x >> 16));
                g[2] = bf16_to_f32((uint16_t)(p.y & 0xffff));
                g[3] = bf16_to_f32((uint16_t)(p.y >> 16));
                g[4] = bf16_to_f32((uint16_t)(p.z & 0xffff));
                g[5] = bf16_to_f32((uint16_t)(p.z >> 16));
                g[6] = bf16_to_f32((uint16_t)(p.w & 0xffff));
                g[7] = bf16_to_f32((uint16_t)(p.w >> 16));
            } else {
#pragma unroll
                for (int r = 0; r < 8; ++r) g[r] = 0.0f;
            }
            float hv[8];
#pragma unroll
            for (int r = 0; r < 8; ++r) {
                float v = part[u][r] + gamma * g[r];
                v = fmaxf(v, 0.0f);
                op[(size_t)(mOff + r) * HWSZ + (size_t)n * WW + j] += v;
                hv[r] = v;
            }
            uint4 p;
            p.x = pk_bf16(hv[0], hv[1]); p.y = pk_bf16(hv[2], hv[3]);
            p.z = pk_bf16(hv[4], hv[5]); p.w = pk_bf16(hv[6], hv[7]);
            *(uint4*)&prevB[n * 72 + mOff] = p;
        }
        __syncthreads();
    }
}

extern "C" void kernel_launch(void* const* d_in, const int* in_sizes, int n_in,
                              void* d_out, int out_size, void* d_ws,
                              size_t ws_size, hipStream_t stream) {
    (void)in_sizes; (void)n_in; (void)ws_size;

    const float* x      = (const float*)d_in[0];
    const float* y      = (const float*)d_in[1];
    const float* Wc     = (const float*)d_in[2];
    const float* bc     = (const float*)d_in[3];
    const float* gammas = (const float*)d_in[4];
    float* out = (float*)d_out;

    float* hs = (float*)d_ws;                       // 32 MB
    float* hn = hs + (size_t)8 * CH * HWSZ;         // 32 MB

    zero_kernel<<<2048, 256, 0, stream>>>(out, out_size);

    // South + north row scans (hs, hn) — 32 independent workgroups.
    row_scan_kernel<<<32, 256, 0, stream>>>(x, y, Wc, bc, hs, hn);

    // Four directional column scans, launched sequentially so the += into
    // d_out is exclusive per launch (deterministic, no fp atomics).
    // (base, gammaIdx, Wt, Wa, Wb, dirSign, shiftDown)
    col_scan_kernel<<<8, 256, 0, stream>>>(hs, y, Wc, bc, gammas, 0,  2,  3,  4, +1, 1, out); // SE
    col_scan_kernel<<<8, 256, 0, stream>>>(hs, y, Wc, bc, gammas, 1,  5,  6,  7, -1, 1, out); // SW
    col_scan_kernel<<<8, 256, 0, stream>>>(hn, y, Wc, bc, gammas, 2, 10, 11, 12, +1, 0, out); // NE
    col_scan_kernel<<<8, 256, 0, stream>>>(hn, y, Wc, bc, gammas, 3, 13, 14, 15, -1, 0, out); // NW
}
